// GCN_62277025792167
// MI455X (gfx1250) — matrix operand order repaired and checked by reading, not compiled, not run on hardware
//
#include <hip/hip_runtime.h>

// ---------------------------------------------------------------------------
// GCN forward for MI455X (gfx1250, wave32).
//   conv(h):  g = dinv ⊙ (h @ W)   -> WMMA bf16 GEMM, fp32 accumulate
//             a = g (self loop);  a[dst] += g[src] over edges (L2 f32 atomics)
//             h' = relu(dinv ⊙ a + b)
// Feature buffers are 12.8 MB each -> the whole working set sits in the
// 192 MB L2, so the scatter phase (~1.3 GB of gather+atomic traffic, the
// roofline limiter) runs at L2 rather than HBM rates.
// GEMM data path: A loaded as 4x global_load_b128 per lane per K-step and
// converted f32->bf16 in-register (v_cvt_pk_bf16_f32); B pre-converted once
// into WMMA register order so each operand is one contiguous 32B per lane.
// ---------------------------------------------------------------------------

typedef __attribute__((ext_vector_type(16))) __bf16 v16bf;
typedef __attribute__((ext_vector_type(8)))  float  v8f;

#define HID 32

// ---------------- degree / normalization ----------------------------------

__global__ void k_deg_init(float* __restrict__ deg, int N) {
    int i = blockIdx.x * blockDim.x + threadIdx.x;
    if (i < N) deg[i] = 1.0f;                       // self-loop contribution
}

__global__ void k_deg_count(const int* __restrict__ ei, float* __restrict__ deg, int E) {
    int e = blockIdx.x * blockDim.x + threadIdx.x;
    if (e < E) unsafeAtomicAdd(&deg[ei[E + e]], 1.0f);   // dst = ei[1][e]
}

__global__ void k_dinv(float* __restrict__ deg, int N) {
    int i = blockIdx.x * blockDim.x + threadIdx.x;
    if (i < N) deg[i] = rsqrtf(deg[i]);             // deg >= 1 always
}

// ---------------- weight pre-pack into WMMA B register order ---------------
// pack index = ((ktile*2 + colhalf)*32 + lane)*16 + slot
// B layout (16-bit 32x16): lane -> column (lane&15) + 16*colhalf,
//                          K = ktile*32 + (slot&8)*2 + (lane>>4)*8 + (slot&7).

__global__ void k_packW(const float* __restrict__ W, __bf16* __restrict__ packW, int K) {
    int t = blockIdx.x * blockDim.x + threadIdx.x;
    int total = (K >> 5) * 1024;
    if (t >= total) return;
    int e    = t & 15;
    int lane = (t >> 4) & 31;
    int op   = (t >> 9) & 1;
    int kt   = t >> 10;
    int k    = kt * 32 + (e & 8) * 2 + (lane >> 4) * 8 + (e & 7);
    int col  = (lane & 15) + op * 16;
    packW[t] = (__bf16)W[k * HID + col];
}

// ---------------- dense GEMM + dinv row-scale (WMMA bf16) ------------------
// One wave computes a 16-row x 32-col output tile: two v_wmma_f32_16x16x32_bf16
// per K-step of 32. Writes the scaled result to BOTH g (scatter source) and
// acc (self-loop initialization of the aggregation buffer).

__global__ __launch_bounds__(256)
void k_gemm_scale(const float* __restrict__ A, const __bf16* __restrict__ packW,
                  const float* __restrict__ dinv,
                  float* __restrict__ g, float* __restrict__ acc,
                  int N, int K, int lda)
{
    const int lane = threadIdx.x & 31;
    const int wave = threadIdx.x >> 5;
    const int rowTile = (blockIdx.x * 8 + wave) * 16;
    if (rowTile >= N) return;                       // wave-uniform: EXEC all 1s

    const int r  = lane & 15;                       // A row in tile / C column
    const int hg = lane >> 4;                       // lane half-group

    v8f c0 = {}; v8f c1 = {};
    for (int k0 = 0; k0 < K; k0 += 32) {
        // A tile 16x32 bf16: lane (hg,r): slots 0..7 -> K=8hg+0..7,
        //                                 slots 8..15 -> K=16+8hg+0..7
        const float* Arow = A + (size_t)(rowTile + r) * lda + k0 + 8 * hg;
        float4 f0 = *(const float4*)(Arow);
        float4 f1 = *(const float4*)(Arow + 4);
        float4 f2 = *(const float4*)(Arow + 16);
        float4 f3 = *(const float4*)(Arow + 20);
        v16bf a;
        a[0]=(__bf16)f0.x;  a[1]=(__bf16)f0.y;  a[2]=(__bf16)f0.z;  a[3]=(__bf16)f0.w;
        a[4]=(__bf16)f1.x;  a[5]=(__bf16)f1.y;  a[6]=(__bf16)f1.z;  a[7]=(__bf16)f1.w;
        a[8]=(__bf16)f2.x;  a[9]=(__bf16)f2.y;  a[10]=(__bf16)f2.z; a[11]=(__bf16)f2.w;
        a[12]=(__bf16)f3.x; a[13]=(__bf16)f3.y; a[14]=(__bf16)f3.z; a[15]=(__bf16)f3.w;

        // B operands: contiguous 32B per lane, pre-packed in register order
        const __bf16* pw = packW + ((size_t)(k0 >> 5) * 1024) + lane * 16;
        v16bf b0 = *(const v16bf*)pw;               // columns 0..15
        v16bf b1 = *(const v16bf*)(pw + 512);       // columns 16..31

        c0 = __builtin_amdgcn_wmma_f32_16x16x32_bf16(false, a, false, b0,
                                                     (short)0, c0, false, false);
        c1 = __builtin_amdgcn_wmma_f32_16x16x32_bf16(false, a, false, b1,
                                                     (short)0, c1, false, false);
    }

    // C/D layout: lane column n = r; VGPR v holds row v + 8*hg
#pragma unroll
    for (int v = 0; v < 8; ++v) {
        int R = rowTile + v + 8 * hg;
        float s = dinv[R];
        float x0 = c0[v] * s;
        float x1 = c1[v] * s;
        size_t o = (size_t)R * HID + r;
        g[o]      = x0;  acc[o]      = x0;
        g[o + 16] = x1;  acc[o + 16] = x1;
    }
}

// ---------------- edge scatter: acc[dst] += g[src] -------------------------
// 8 lanes per edge, float4 each -> one contiguous 128B row per edge for both
// the gather and the atomic adds. unsafeAtomicAdd -> global_atomic_add_f32.

__global__ __launch_bounds__(256)
void k_scatter(const int* __restrict__ ei, const float* __restrict__ g,
               float* __restrict__ acc, int E)
{
    long long t = (long long)blockIdx.x * blockDim.x + threadIdx.x;
    int e = (int)(t >> 3);
    if (e >= E) return;
    int c = ((int)t & 7) * 4;
    int src = ei[e];
    int dst = ei[E + e];
    const float4 v = *(const float4*)(g + (size_t)src * HID + c);
    float* d = acc + (size_t)dst * HID + c;
    unsafeAtomicAdd(d + 0, v.x);
    unsafeAtomicAdd(d + 1, v.y);
    unsafeAtomicAdd(d + 2, v.z);
    unsafeAtomicAdd(d + 3, v.w);
}

// ---------------- epilogue: h = relu(dinv ⊙ acc + b) -----------------------

__global__ __launch_bounds__(256)
void k_finish(const float* __restrict__ acc, const float* __restrict__ dinv,
              const float* __restrict__ b, float* __restrict__ h, int N)
{
    long long t = (long long)blockIdx.x * blockDim.x + threadIdx.x;
    int row = (int)(t >> 3);
    if (row >= N) return;
    int c = ((int)t & 7) * 4;
    float s   = dinv[row];
    float4 v  = *(const float4*)(acc + (size_t)row * HID + c);
    float4 bb = *(const float4*)(b + c);
    float4 o;
    o.x = fmaxf(fmaf(v.x, s, bb.x), 0.0f);
    o.y = fmaxf(fmaf(v.y, s, bb.y), 0.0f);
    o.z = fmaxf(fmaf(v.z, s, bb.z), 0.0f);
    o.w = fmaxf(fmaf(v.w, s, bb.w), 0.0f);
    *(float4*)(h + (size_t)row * HID + c) = o;
}

// ---------------- final FC: concat(h[0],h[1]) @ Wfc + bfc ------------------
// concat(h[0],h[1]) is exactly h[0..63] in row-major N x 32 storage.

__global__ void k_fc(const float* __restrict__ h, const float* __restrict__ Wfc,
                     const float* __restrict__ bfc, float* __restrict__ out)
{
    int o = threadIdx.x;                            // 64 outputs
    float s = bfc[o];
#pragma unroll 4
    for (int k = 0; k < 64; ++k)
        s = fmaf(h[k], Wfc[k * 64 + o], s);
    out[o] = s;
}

// ---------------------------------------------------------------------------

extern "C" void kernel_launch(void* const* d_in, const int* in_sizes, int n_in,
                              void* d_out, int out_size, void* d_ws, size_t ws_size,
                              hipStream_t stream)
{
    const float* x    = (const float*)d_in[0];   // [N,128]
    const int*   ei   = (const int*)  d_in[1];   // [2,E]
    const float* W1   = (const float*)d_in[2];
    const float* b1   = (const float*)d_in[3];
    const float* W2   = (const float*)d_in[4];
    const float* b2   = (const float*)d_in[5];
    const float* W3   = (const float*)d_in[6];
    const float* b3   = (const float*)d_in[7];
    const float* Wfc  = (const float*)d_in[8];
    const float* bfc  = (const float*)d_in[9];
    float* out = (float*)d_out;

    const int N = in_sizes[0] / 128;
    const int E = in_sizes[1] / 2;

    // workspace layout (floats): dinv | P (g) | A (acc) | Q (h) | packed weights
    float* dinv = (float*)d_ws;
    float* P  = dinv + 102400;                      // N rounded up, 32B aligned
    float* Aw = P  + (size_t)N * HID;
    float* Q  = Aw + (size_t)N * HID;
    __bf16* pw1 = (__bf16*)(Q + (size_t)N * HID);   // 4 ktiles * 1024 = 4096 bf16
    __bf16* pw2 = pw1 + 4096;                       // 1024 bf16
    __bf16* pw3 = pw2 + 1024;                       // 1024 bf16

    const int TB = 256;
    dim3 blk(TB);
    int gN    = (N + TB - 1) / TB;
    int gE    = (E + TB - 1) / TB;
    int gGemm = ((N / 16) + 7) / 8;                 // 8 waves (16 rows each) per block
    long long sc = (long long)E * 8;
    int gScat = (int)((sc + TB - 1) / TB);
    long long fn = (long long)N * 8;
    int gFin  = (int)((fn + TB - 1) / TB);

    // --- weight pre-pack (tiny, once per call) ---
    k_packW<<<(4096 + TB - 1) / TB, blk, 0, stream>>>(W1, pw1, 128);
    k_packW<<<(1024 + TB - 1) / TB, blk, 0, stream>>>(W2, pw2, 32);
    k_packW<<<(1024 + TB - 1) / TB, blk, 0, stream>>>(W3, pw3, 32);

    // --- normalization ---
    k_deg_init <<<gN, blk, 0, stream>>>(dinv, N);
    k_deg_count<<<gE, blk, 0, stream>>>(ei, dinv, E);
    k_dinv     <<<gN, blk, 0, stream>>>(dinv, N);

    // --- layer 1: x[N,128] ---
    k_gemm_scale<<<gGemm, blk, 0, stream>>>(x, pw1, dinv, P, Aw, N, 128, 128);
    k_scatter   <<<gScat, blk, 0, stream>>>(ei, P, Aw, E);
    k_finish    <<<gFin,  blk, 0, stream>>>(Aw, dinv, b1, Q, N);

    // --- layer 2 ---
    k_gemm_scale<<<gGemm, blk, 0, stream>>>(Q, pw2, dinv, P, Aw, N, 32, 32);
    k_scatter   <<<gScat, blk, 0, stream>>>(ei, P, Aw, E);
    k_finish    <<<gFin,  blk, 0, stream>>>(Aw, dinv, b2, Q, N);

    // --- layer 3 ---
    k_gemm_scale<<<gGemm, blk, 0, stream>>>(Q, pw3, dinv, P, Aw, N, 32, 32);
    k_scatter   <<<gScat, blk, 0, stream>>>(ei, P, Aw, E);
    k_finish    <<<gFin,  blk, 0, stream>>>(Aw, dinv, b3, Q, N);

    // --- FC head ---
    k_fc<<<1, 64, 0, stream>>>(Q, Wfc, bfc, out);
}
